// DeepSetBaseline_35828617183720
// MI455X (gfx1250) — compile-verified, matrix operand order checked by reading, not run on hardware
//
#include <hip/hip_runtime.h>
#include <cmath>

typedef __attribute__((ext_vector_type(16))) _Float16 v16h;
typedef __attribute__((ext_vector_type(8)))  _Float16 v8h;
typedef __attribute__((ext_vector_type(8)))  float    v8f;

#define KDIM 512          // inner dim of every GEMM
#define NKT  16           // 512 / 32 K-steps
#define WELEMS (512*512)  // elements per weight matrix
#define MROWS 32          // rows per block (2 WMMA M-tiles)

// ---- WMMA fragment loaders -------------------------------------------------
// A-matrix 16x32 f16 layout (wave32): lane L holds row M=L%16;
// halves 0..7  = K: k0 + (L>=16 ? 8 : 0) + 0..7
// halves 8..15 = K: k0 + 16 + (L>=16 ? 8 : 0) + 0..7
__device__ __forceinline__ v16h load_a_lds(const _Float16* buf, int lane, int kt) {
    int r   = lane & 15;
    int off = (lane >> 4) << 3;
    const _Float16* p = buf + r * KDIM + kt * 32 + off;
    union { v16h v; v8h h[2]; } u;
    u.h[0] = *(const v8h*)p;
    u.h[1] = *(const v8h*)(p + 16);
    return u.v;
}

// B-matrix fragments pre-packed in ws: tile (ntg, kt) stored as 32 lanes x 16
// contiguous halves -> one 32-byte load per lane.
__device__ __forceinline__ v16h load_b_packed(const _Float16* wp, int lane, int kt, int ntg) {
    return *(const v16h*)(wp + (size_t)((((ntg << 4) + kt) << 5) + lane) * 16);
}

// ---- one dense layer over MT 16-row tiles: out = tanh(in @ W + b) ----------
// Each B fragment is reused for MT WMMAs (register blocking on M).
template<int MT>
__device__ __forceinline__ void gemm_layer_tanh(
    const _Float16* __restrict__ bufIn, _Float16* __restrict__ bufOut,
    const _Float16* __restrict__ Wp, const float* __restrict__ bias,
    int wave, int lane) {
    v8f acc[MT][4] = {};
    for (int kt = 0; kt < NKT; ++kt) {
        v16h a[MT];
#pragma unroll
        for (int m = 0; m < MT; ++m)
            a[m] = load_a_lds(bufIn + m * 16 * KDIM, lane, kt);
#pragma unroll
        for (int nt = 0; nt < 4; ++nt) {
            v16h bm = load_b_packed(Wp, lane, kt, wave * 4 + nt);
#pragma unroll
            for (int m = 0; m < MT; ++m)
                acc[m][nt] = __builtin_amdgcn_wmma_f32_16x16x32_f16(
                    false, a[m], false, bm, (short)0, acc[m][nt], false, false);
        }
    }
    int hi = (lane >> 4) << 3;   // row offset within C/D tile
    int nl = lane & 15;
#pragma unroll
    for (int m = 0; m < MT; ++m) {
#pragma unroll
        for (int nt = 0; nt < 4; ++nt) {
            int N = wave * 64 + nt * 16 + nl;
            float bv = bias[N];
#pragma unroll
            for (int i = 0; i < 8; ++i) {
                float v = tanhf(acc[m][nt][i] + bv);
                bufOut[(m * 16 + i + hi) * KDIM + N] = (_Float16)v;
            }
        }
    }
}

// ---- final encoder layer: enc = in @ W3 + b3, mask rows, reduce over M,
// atomic-add into codes[b] ---------------------------------------------------
template<int MT>
__device__ __forceinline__ void gemm_layer_mask_reduce(
    const _Float16* __restrict__ bufIn,
    const _Float16* __restrict__ Wp, const float* __restrict__ bias,
    const float* __restrict__ rowsum, float* __restrict__ codesRow,
    int wave, int lane) {
    v8f acc[MT][4] = {};
    for (int kt = 0; kt < NKT; ++kt) {
        v16h a[MT];
#pragma unroll
        for (int m = 0; m < MT; ++m)
            a[m] = load_a_lds(bufIn + m * 16 * KDIM, lane, kt);
#pragma unroll
        for (int nt = 0; nt < 4; ++nt) {
            v16h bm = load_b_packed(Wp, lane, kt, wave * 4 + nt);
#pragma unroll
            for (int m = 0; m < MT; ++m)
                acc[m][nt] = __builtin_amdgcn_wmma_f32_16x16x32_f16(
                    false, a[m], false, bm, (short)0, acc[m][nt], false, false);
        }
    }
    int hi = (lane >> 4) << 3;
    int nl = lane & 15;
#pragma unroll
    for (int nt = 0; nt < 4; ++nt) {
        int N = wave * 64 + nt * 16 + nl;
        float bv = bias[N];
        float s = 0.f;
#pragma unroll
        for (int m = 0; m < MT; ++m) {
#pragma unroll
            for (int i = 0; i < 8; ++i) {
                float mk = (rowsum[m * 16 + i + hi] != 0.0f) ? 1.0f : 0.0f;
                s += (acc[m][nt][i] + bv) * mk;   // sign(|sum|) row mask
            }
        }
        unsafeAtomicAdd(&codesRow[N], s);   // global_atomic_add_f32
    }
}

// ---- fp32 tile -> f16 LDS loader (optionally accumulating row sums) --------
__device__ __forceinline__ void load_tile_f16(
    const float* __restrict__ src, _Float16* __restrict__ dst,
    float* __restrict__ rowsum, int t) {
    int r  = t >> 3;             // 0..31
    int c0 = (t & 7) * 64;       // 64 cols per thread
    const float* xr = src + r * KDIM + c0;
    float s = 0.0f;
    for (int j = 0; j < 64; j += 8) {
        float4 a = *(const float4*)(xr + j);
        float4 c = *(const float4*)(xr + j + 4);
        s += a.x + a.y + a.z + a.w + c.x + c.y + c.z + c.w;
        v8h h;
        h[0] = (_Float16)a.x; h[1] = (_Float16)a.y;
        h[2] = (_Float16)a.z; h[3] = (_Float16)a.w;
        h[4] = (_Float16)c.x; h[5] = (_Float16)c.y;
        h[6] = (_Float16)c.z; h[7] = (_Float16)c.w;
        *(v8h*)(dst + r * KDIM + c0 + j) = h;
    }
    if (rowsum) atomicAdd(&rowsum[r], s);   // ds_add_f32
}

// ---- kernel 0: pack W1..W5 to f16 B-fragment layout, zero codes ------------
__global__ __launch_bounds__(256) void prep_kernel(
    const float* __restrict__ W1, const float* __restrict__ W2,
    const float* __restrict__ W3, const float* __restrict__ W4,
    const float* __restrict__ W5,
    _Float16* __restrict__ wsW, float* __restrict__ codes) {
    int idx = blockIdx.x * blockDim.x + threadIdx.x;
    if (idx < 64 * 512) codes[idx] = 0.0f;
    if (idx >= 5 * WELEMS) return;
    int wi = idx / WELEMS, e = idx % WELEMS;
    const float* src = (wi == 0) ? W1 : (wi == 1) ? W2 : (wi == 2) ? W3
                     : (wi == 3) ? W4 : W5;
    int k = e >> 9, n = e & 511;                 // W is [K=fan_in][N=fan_out]
    int kt = k >> 5, kk = k & 31;
    int ntg = n >> 4, nl = n & 15;
    int lane = nl + (((kk >> 3) & 1) << 4);
    int half = (kk & 7) + ((kk >> 4) << 3);
    wsW[(size_t)wi * WELEMS +
        (size_t)((((ntg << 4) + kt) << 5) + lane) * 16 + half] = (_Float16)src[e];
}

// ---- kernel 1: fused 3-layer encoder + masked segment-sum ------------------
__global__ __launch_bounds__(256) void encoder_kernel(
    const float* __restrict__ words,
    const _Float16* __restrict__ W1p, const float* __restrict__ b1,
    const _Float16* __restrict__ W2p, const float* __restrict__ b2,
    const _Float16* __restrict__ W3p, const float* __restrict__ b3,
    float* __restrict__ codes) {
    __shared__ __align__(16) _Float16 bufA[MROWS * KDIM];
    __shared__ __align__(16) _Float16 bufB[MROWS * KDIM];
    __shared__ float rowsum[MROWS];

    int t  = threadIdx.x;
    int b  = blockIdx.x >> 5;    // batch (1024/32 = 32 tiles per batch)
    int mt = blockIdx.x & 31;    // 32-row tile within S=1024
    const float* X = words + ((size_t)b * 1024 + (size_t)mt * MROWS) * KDIM;

    if (t < MROWS) rowsum[t] = 0.0f;
    __syncthreads();

    load_tile_f16(X, bufA, rowsum, t);
    __syncthreads();

    int wave = t >> 5, lane = t & 31;
    gemm_layer_tanh<2>(bufA, bufB, W1p, b1, wave, lane);
    __syncthreads();
    gemm_layer_tanh<2>(bufB, bufA, W2p, b2, wave, lane);
    __syncthreads();
    gemm_layer_mask_reduce<2>(bufA, W3p, b3, rowsum, codes + (size_t)b * 512,
                              wave, lane);
}

// ---- kernel 2: decoder on codes[64,512] ------------------------------------
__global__ __launch_bounds__(256) void decoder_kernel(
    const float* __restrict__ codes,
    const _Float16* __restrict__ W4p, const float* __restrict__ b4,
    const _Float16* __restrict__ W5p, const float* __restrict__ b5,
    const float* __restrict__ W6, const float* __restrict__ b6,
    float* __restrict__ out) {
    __shared__ __align__(16) _Float16 bufA[MROWS * KDIM];
    __shared__ __align__(16) _Float16 bufB[MROWS * KDIM];
    __shared__ float osum[MROWS];

    int t = threadIdx.x;
    const float* C = codes + (size_t)blockIdx.x * MROWS * KDIM;
    if (t < MROWS) osum[t] = 0.0f;

    load_tile_f16(C, bufA, nullptr, t);
    __syncthreads();

    int wave = t >> 5, lane = t & 31;
    gemm_layer_tanh<2>(bufA, bufB, W4p, b4, wave, lane);
    __syncthreads();
    gemm_layer_tanh<2>(bufB, bufA, W5p, b5, wave, lane);
    __syncthreads();

    // final 512 -> 1 projection: scalar dot per row
    {
        int r  = t >> 3;
        int c0 = (t & 7) * 64;
        float s = 0.0f;
        for (int j = 0; j < 64; ++j)
            s += (float)bufA[r * KDIM + c0 + j] * W6[c0 + j];
        atomicAdd(&osum[r], s);
    }
    __syncthreads();
    if (t < MROWS) out[blockIdx.x * MROWS + t] = osum[t] + b6[0];
}

// ---- host entry ------------------------------------------------------------
extern "C" void kernel_launch(void* const* d_in, const int* in_sizes, int n_in,
                              void* d_out, int out_size, void* d_ws, size_t ws_size,
                              hipStream_t stream) {
    const float* words = (const float*)d_in[0];
    const float* W1 = (const float*)d_in[1];  const float* b1 = (const float*)d_in[2];
    const float* W2 = (const float*)d_in[3];  const float* b2 = (const float*)d_in[4];
    const float* W3 = (const float*)d_in[5];  const float* b3 = (const float*)d_in[6];
    const float* W4 = (const float*)d_in[7];  const float* b4 = (const float*)d_in[8];
    const float* W5 = (const float*)d_in[9];  const float* b5 = (const float*)d_in[10];
    const float* W6 = (const float*)d_in[11]; const float* b6 = (const float*)d_in[12];

    _Float16* wsW = (_Float16*)d_ws;                          // 5 x 512KB f16 packed
    float* codes  = (float*)((char*)d_ws + (size_t)5 * WELEMS * sizeof(_Float16));
    float* out    = (float*)d_out;

    int prepN = 5 * WELEMS;
    prep_kernel<<<(prepN + 255) / 256, 256, 0, stream>>>(W1, W2, W3, W4, W5, wsW, codes);

    encoder_kernel<<<64 * 32, 256, 0, stream>>>(
        words, wsW, b1, wsW + 1 * WELEMS, b2, wsW + 2 * WELEMS, b3, codes);

    decoder_kernel<<<2, 256, 0, stream>>>(
        codes, wsW + 3 * WELEMS, b4, wsW + 4 * WELEMS, b5, W6, b6, out);
}
